// DOSTransformer_phonon_77824807403586
// MI455X (gfx1250) — compile-verified, matrix operand order checked by reading, not run on hardware
//
#include <hip/hip_runtime.h>
#include <hip/hip_bf16.h>
#include <math.h>

typedef float v2f  __attribute__((ext_vector_type(2)));
typedef float v8f  __attribute__((ext_vector_type(8)));
typedef _Float16 v16h __attribute__((ext_vector_type(16)));

#if defined(__has_builtin)
#if __has_builtin(__builtin_amdgcn_wmma_f32_16x16x4_f32)
#define HAVE_WMMA_F32 1
#endif
#endif

#define C_H     128
#define C_H2    64
#define C_B     64
#define C_NPG   256
#define C_N     (C_B * C_NPG)      // 16384
#define C_DEG   24
#define C_E     (C_N * C_DEG)      // 393216
#define C_NATOM 118
#define C_NE    51
#define C_FFN   512
#define CH_EDGE 65536
#define N_CHUNK (C_E / CH_EDGE)    // 6

#define TM 32
#define TN 64
#define KBK 32

// ---------------------------------------------------------------- GEMM params
struct GemmP {
  const float* A;   long lda;   long bsA;
  const float* Bw;  long ldb;   long bsB;  int transB;
  const float* bias;
  float* C;         long ldc;   long bsC;
  const float* Res; long ldres; long bsRes;
  int M, Nn, K;
  float scale, alpha;
  const float* aP;                 // device prelu alpha (optional)
  // AMODE 1: concat3 h[row],h[col],e ; AMODE 2: concat2 g0[row],g1[row]
  const float* g0; const float* g1; const float* g2;
  const int* rowIdx; const int* colIdx; int rowBase;
  // EPI 1: edge epilogue (e += ne ; atomic agg)
  float* eAcc; float* agg;
};

// ---------------------------------------------------------------- GEMM kernel
// 128 threads = 4 waves; block tile 32(M) x 64(N); per-wave 16x32 (two WMMA
// accumulators sharing one A fragment).  Global->LDS staging is software
// pipelined through registers so loads overlap the WMMA stream.
template<int ACT, int AMODE, int EPI>
__global__ __launch_bounds__(128) void gemm_wmma(GemmP p) {
  __shared__ float As[TM][KBK + 1];
  __shared__ float Bs[KBK][TN + 1];
  const int tid  = threadIdx.x;
  const int wave = tid >> 5;
  const int lane = tid & 31;
  const int wr = (wave >> 1) * 16;
  const int wc = (wave & 1) * 32;
  const int bm = blockIdx.y * TM;
  const int bn = blockIdx.x * TN;
  const int bz = blockIdx.z;
  const float* Ab = p.A  + (long)bz * p.bsA;
  const float* Bb = p.Bw + (long)bz * p.bsB;
  float*       Cb = p.C  + (long)bz * p.bsC;

  v8f acc0 = {};
  v8f acc1 = {};
  const int half = lane >> 4;
  const int ml   = lane & 15;

  float ra[8];    // staged A slab (32x32 / 128 thr)
  float rb[16];   // staged B slab (32x64 / 128 thr)

  auto loadA = [&](int k0) {
    #pragma unroll
    for (int i = 0; i < 8; ++i) {
      int t = tid + i * 128;
      int r = t >> 5, c = t & 31;
      int gr = bm + r, gk = k0 + c;
      float v = 0.f;
      if (gr < p.M && gk < p.K) {
        if (AMODE == 0) {
          v = Ab[(long)gr * p.lda + gk];
        } else if (AMODE == 1) {
          long ge = (long)p.rowBase + gr;
          if (gk < C_H)            v = p.g0[(long)p.rowIdx[ge] * C_H + gk];
          else if (gk < 2 * C_H)   v = p.g0[(long)p.colIdx[ge] * C_H + (gk - C_H)];
          else                     v = p.g2[ge * C_H + (gk - 2 * C_H)];
        } else {
          if (gk < C_H)            v = p.g0[(long)gr * C_H + gk];
          else                     v = p.g1[(long)gr * C_H + (gk - C_H)];
        }
      }
      ra[i] = v;
    }
  };
  auto loadB = [&](int k0) {
    #pragma unroll
    for (int i = 0; i < 16; ++i) {
      int t = tid + i * 128;
      int r = t >> 6, c = t & 63;
      int gk = k0 + r, gn = bn + c;
      float v = 0.f;
      if (gk < p.K && gn < p.Nn)
        v = p.transB ? Bb[(long)gn * p.ldb + gk] : Bb[(long)gk * p.ldb + gn];
      rb[i] = v;
    }
  };
  auto commit = [&]() {
    #pragma unroll
    for (int i = 0; i < 8; ++i) {
      int t = tid + i * 128;
      As[t >> 5][t & 31] = ra[i];
    }
    #pragma unroll
    for (int i = 0; i < 16; ++i) {
      int t = tid + i * 128;
      Bs[t >> 6][t & 63] = rb[i];
    }
  };

  loadA(0);
  loadB(0);
  commit();
  __syncthreads();

  for (int k0 = 0; k0 < p.K; k0 += KBK) {
    const bool more = (k0 + KBK) < p.K;
    if (more) {
      if (AMODE == 0 && k0 + 2 * KBK < p.K) {
        // touch the slab after next (global_prefetch_b8)
        __builtin_prefetch(&Ab[(long)(bm + (tid & 31)) * p.lda + k0 + 2 * KBK], 0, 0);
        __builtin_prefetch(&Bb[(long)(k0 + 2 * KBK) * (p.transB ? 1 : p.ldb) + bn], 0, 0);
      }
      loadA(k0 + KBK);            // global loads overlap the WMMAs below
      loadB(k0 + KBK);
    }
#if HAVE_WMMA_F32
    #pragma unroll
    for (int kk = 0; kk < KBK; kk += 4) {
      v2f a, b0, b1;
      a.x  = As[wr + ml][kk + half * 2 + 0];
      a.y  = As[wr + ml][kk + half * 2 + 1];
      b0.x = Bs[kk + half * 2 + 0][wc + ml];
      b0.y = Bs[kk + half * 2 + 1][wc + ml];
      b1.x = Bs[kk + half * 2 + 0][wc + 16 + ml];
      b1.y = Bs[kk + half * 2 + 1][wc + 16 + ml];
      acc0 = __builtin_amdgcn_wmma_f32_16x16x4_f32(false, a, false, b0, (short)0,
                                                   acc0, false, false);
      acc1 = __builtin_amdgcn_wmma_f32_16x16x4_f32(false, a, false, b1, (short)0,
                                                   acc1, false, false);
    }
#else
    {
      v16h a, b0, b1;
      #pragma unroll
      for (int e2 = 0; e2 < 16; ++e2) {
        int kdx = (e2 < 8) ? (e2 + half * 8) : ((e2 - 8) + 16 + half * 8);
        a[e2]  = (_Float16)As[wr + ml][kdx];
        b0[e2] = (_Float16)Bs[kdx][wc + ml];
        b1[e2] = (_Float16)Bs[kdx][wc + 16 + ml];
      }
      acc0 = __builtin_amdgcn_wmma_f32_16x16x32_f16(false, a, false, b0, (short)0,
                                                    acc0, false, false);
      acc1 = __builtin_amdgcn_wmma_f32_16x16x32_f16(false, a, false, b1, (short)0,
                                                    acc1, false, false);
    }
#endif
    __syncthreads();
    if (more) {
      commit();
      __syncthreads();
    }
  }

  // epilogue
  const int col0 = bn + wc + ml;
  const int col1 = col0 + 16;
  const int rbase = bm + wr + (half ? 8 : 0);
  float al = (ACT == 2) ? (p.aP ? p.aP[0] : p.alpha) : p.alpha;
  #pragma unroll
  for (int cs = 0; cs < 2; ++cs) {
    int col = cs ? col1 : col0;
    if (col >= p.Nn) continue;
    float bv = p.bias ? p.bias[col] : 0.f;
    #pragma unroll
    for (int j = 0; j < 8; ++j) {
      int row = rbase + j;
      if (row < p.M) {
        float v = ((cs ? acc1[j] : acc0[j]) + bv) * p.scale;
        if      (ACT == 1) v = fmaxf(v, 0.f);
        else if (ACT == 2) v = v > 0.f ? v : al * v;
        else if (ACT == 3) v = v > 0.f ? v : 0.01f * v;
        if (EPI == 0) {
          if (p.Res) v += p.Res[(long)bz * p.bsRes + (long)row * p.ldres + col];
          Cb[(long)row * p.ldc + col] = v;
        } else {
          long ge = (long)p.rowBase + row;
          p.eAcc[ge * C_H + col] += v;                                    // e += ne
          atomicAdd(&p.agg[(long)p.colIdx[ge] * C_H + col], v);           // scatter
        }
      }
    }
  }
}

template<int ACT, int AMODE, int EPI>
static void launch_gemm(const GemmP& p, int batch, hipStream_t st) {
  dim3 g((unsigned)((p.Nn + TN - 1) / TN), (unsigned)((p.M + TM - 1) / TM),
         (unsigned)batch);
  gemm_wmma<ACT, AMODE, EPI><<<g, dim3(128), 0, st>>>(p);
}

// ------------------------------------------------------------ small kernels
__global__ void fill_f32(float* p, float v, long n) {
  long i = (long)blockIdx.x * blockDim.x + threadIdx.x;
  if (i < n) p[i] = v;
}
__global__ void copy_f32(const float* s, float* d, long n) {
  long i = (long)blockIdx.x * blockDim.x + threadIdx.x;
  if (i < n) d[i] = s[i];
}
__global__ void edge_attr_k(const float* ev, float* ea, long nE) {
  long i = (long)blockIdx.x * blockDim.x + threadIdx.x;
  if (i >= nE) return;
  float x = ev[3 * i], y = ev[3 * i + 1], z = ev[3 * i + 2];
  float r = sqrtf(x * x + y * y + z * z);
  float inv = 1.f / fmaxf(r, 1e-12f);
  float u = 2.f * (r * 0.25f - 1.f);
  float cut = (u > 0.f) ? 0.f
            : (u < -1.f) ? 1.f
            : 0.5f * (1.f - cosf(3.14159265358979f * u));
  float s3 = 1.7320508075688772f;
  ea[4 * i + 0] = cut;
  ea[4 * i + 1] = cut * s3 * x * inv;
  ea[4 * i + 2] = cut * s3 * y * inv;
  ea[4 * i + 3] = cut * s3 * z * inv;
}
__global__ void count_k(const int* col, float* cnt, long nE) {
  long i = (long)blockIdx.x * blockDim.x + threadIdx.x;
  if (i < nE) atomicAdd(&cnt[col[i]], 1.f);
}
__global__ void divide_k(float* agg, const float* cnt, long n) {
  long i = (long)blockIdx.x * blockDim.x + threadIdx.x;
  if (i < n) agg[i] /= fmaxf(cnt[i / C_H], 1.f);
}
__global__ void ln_act(const float* x, const float* g, const float* b,
                       const float* aP, float* y, int D, int act, long rows) {
  long row = blockIdx.x;
  if (row >= rows) return;
  const float* xr = x + row * D;
  float* yr = y + row * D;
  __shared__ float ssum[128], ssq[128];
  int t = threadIdx.x;
  float s = 0.f, q = 0.f;
  for (int c = t; c < D; c += 128) { float v = xr[c]; s += v; q += v * v; }
  ssum[t] = s; ssq[t] = q; __syncthreads();
  for (int off = 64; off > 0; off >>= 1) {
    if (t < off) { ssum[t] += ssum[t + off]; ssq[t] += ssq[t + off]; }
    __syncthreads();
  }
  float mean = ssum[0] / D;
  float var  = ssq[0] / D - mean * mean;
  float inv  = rsqrtf(var + 1e-5f);
  float al   = aP ? aP[0] : 0.25f;
  for (int c = t; c < D; c += 128) {
    float v = (xr[c] - mean) * inv * g[c] + b[c];
    if (act == 2) v = v > 0.f ? v : al * v;
    yr[c] = v;
  }
}
__global__ void softmax_rows(float* x, int T) {
  long row = blockIdx.x;
  float* xr = x + row * T;
  __shared__ float red[256];
  int t = threadIdx.x;
  float v = (t < T) ? xr[t] : -3.0e38f;
  red[t] = v; __syncthreads();
  for (int off = 128; off > 0; off >>= 1) {
    if (t < off) red[t] = fmaxf(red[t], red[t + off]);
    __syncthreads();
  }
  float m = red[0]; __syncthreads();
  float e = (t < T) ? __expf(v - m) : 0.f;
  red[t] = e; __syncthreads();
  for (int off = 128; off > 0; off >>= 1) {
    if (t < off) red[t] += red[t + off];
    __syncthreads();
  }
  if (t < T) xr[t] = e / red[0];
}
__global__ void posadd_k(const float* xin, float* xout, float scale, long n) {
  long i = (long)blockIdx.x * blockDim.x + threadIdx.x;
  if (i >= n) return;
  int f = (int)(i % C_H);
  long s = i / ((long)C_B * C_H);
  int j = (f < C_H2) ? f : f - C_H2;
  float freq = __expf(-(logf(10000.f) / (C_H2 - 1)) * (float)j);
  float ang = (float)s * freq;
  float pv = (f < C_H2) ? __sinf(ang) : __cosf(ang);
  xout[i] = scale * xin[i] + pv;
}
__global__ void xdense_k(const float* h, float* xd, long n) {
  long i = (long)blockIdx.x * blockDim.x + threadIdx.x;
  if (i >= n) return;
  int f = (int)(i % C_H);
  long node = i / C_H;
  int b = (int)(node / C_NPG), pp = (int)(node % C_NPG);
  xd[((long)pp * C_B + b) * C_H + f] = h[i];
}
__global__ void bcast_emb_k(const float* emb, float* out, long n) {
  long i = (long)blockIdx.x * blockDim.x + threadIdx.x;
  if (i >= n) return;
  int f = (int)(i % C_H);
  int e = (int)(i / ((long)C_B * C_H));
  out[i] = emb[(long)e * C_H + f];
}
__global__ void graphsum_k(const float* h, float* gs) {
  int g = blockIdx.x, f = threadIdx.x;
  float s = 0.f;
  for (int p = 0; p < C_NPG; ++p) s += h[((long)g * C_NPG + p) * C_H + f];
  gs[(long)g * C_H + f] = s;
}
__global__ void cat_eg_k(const float* energ, const float* gv, float* out, long n) {
  long i = (long)blockIdx.x * blockDim.x + threadIdx.x;
  if (i >= n) return;
  long r = i / (2 * C_H); int k = (int)(i % (2 * C_H));
  int b = (int)(r % C_B);
  out[i] = (k < C_H) ? energ[r * C_H + k] : gv[(long)b * C_H + (k - C_H)];
}
__global__ void cat_egp_k(const float* energ, const float* gv, const float* prompt,
                          const int* sys, float* out, long n) {
  long i = (long)blockIdx.x * blockDim.x + threadIdx.x;
  if (i >= n) return;
  long r = i / (2 * C_H + C_H2); int k = (int)(i % (2 * C_H + C_H2));
  int b = (int)(r % C_B);
  float v;
  if (k < C_H)            v = energ[r * C_H + k];
  else if (k < 2 * C_H)   v = gv[(long)b * C_H + (k - C_H)];
  else                    v = prompt[(long)sys[b] * C_H2 + (k - 2 * C_H)];
  out[i] = v;
}
__global__ void outproj_k(const float* dg, const float* w, const float* bb,
                          float* dos) {
  int i = blockIdx.x * blockDim.x + threadIdx.x;
  if (i >= C_NE * C_B) return;
  int e = i / C_B, b = i % C_B;
  const float* r = dg + (long)i * C_H;
  float s = 0.f;
  for (int f = 0; f < C_H; ++f) s += r[f] * w[f];
  dos[(long)b * C_NE + e] = s + bb[0];
}

// ------------------------------------------------------------ host helpers
static inline const float* PF(void* const* d, int i) { return (const float*)d[i]; }
static inline const int*   PI(void* const* d, int i) { return (const int*)d[i]; }

static inline GemmP mkG(const float* A, long lda, const float* W, long ldb,
                        const float* bias, float* C, long ldc,
                        int M, int Nn, int K) {
  GemmP p{};
  p.A = A; p.lda = lda; p.Bw = W; p.ldb = ldb; p.bias = bias;
  p.C = C; p.ldc = ldc; p.M = M; p.Nn = Nn; p.K = K;
  p.scale = 1.f; p.alpha = 0.25f;
  return p;
}
#define EL(n) dim3((unsigned)(((n) + 255) / 256)), dim3(256)

static void run_ln(const float* x, const float* g, const float* b, const float* aP,
                   float* y, int D, int act, long rows, hipStream_t st) {
  ln_act<<<dim3((unsigned)rows), dim3(128), 0, st>>>(x, g, b, aP, y, D, act, rows);
}

// Transformer encoder (TLAYERS=2).  tb = index of first leaf of tenc.
static void run_transformer(void* const* din, int tb, int S, int T,
                            const float* xq, const float* xk, const float* xv,
                            float* out,
                            float* tx, float* txn, float* tq, float* tctx,
                            float* tffh, float* tk, float* tv, float* tkn,
                            float* tvn, float* tkk, float* tvv, float* tsc,
                            hipStream_t st) {
  const long SB = (long)S * C_B, TB = (long)T * C_B;
  const float escale = sqrtf((float)C_H);
  posadd_k<<<EL(SB * C_H), 0, st>>>(xq, tx, escale, SB * C_H);
  posadd_k<<<EL(TB * C_H), 0, st>>>(xk, tk, escale, TB * C_H);
  posadd_k<<<EL(TB * C_H), 0, st>>>(xv, tv, escale, TB * C_H);
  for (int l = 0; l < 2; ++l) {
    int L = tb + l * 16;
    run_ln(tx, PF(din, L + 8), PF(din, L + 9), nullptr, txn, C_H, 0, SB, st);
    run_ln(tk, PF(din, L + 8), PF(din, L + 9), nullptr, tkn, C_H, 0, TB, st);
    run_ln(tv, PF(din, L + 8), PF(din, L + 9), nullptr, tvn, C_H, 0, TB, st);
    { GemmP p = mkG(txn, C_H, PF(din, L + 0), C_H, PF(din, L + 1), tq, C_H,
                    (int)SB, C_H, C_H);
      p.scale = 1.f / escale;
      launch_gemm<0, 0, 0>(p, 1, st); }
    { GemmP p = mkG(tkn, C_H, PF(din, L + 2), C_H, PF(din, L + 3), tkk, C_H,
                    (int)TB, C_H, C_H);
      launch_gemm<0, 0, 0>(p, 1, st); }
    { GemmP p = mkG(tvn, C_H, PF(din, L + 4), C_H, PF(din, L + 5), tvv, C_H,
                    (int)TB, C_H, C_H);
      launch_gemm<0, 0, 0>(p, 1, st); }
    { GemmP p{};                                  // scores[b,s,t] = q . k
      p.A = tq; p.lda = (long)C_B * C_H; p.bsA = C_H;
      p.Bw = tkk; p.ldb = (long)C_B * C_H; p.bsB = C_H; p.transB = 1;
      p.C = tsc; p.ldc = T; p.bsC = (long)S * T;
      p.M = S; p.Nn = T; p.K = C_H; p.scale = 1.f;
      launch_gemm<0, 0, 0>(p, C_B, st); }
    softmax_rows<<<dim3((unsigned)(C_B * S)), dim3(256), 0, st>>>(tsc, T);
    { GemmP p{};                                  // ctx[s,b,h] = att @ v
      p.A = tsc; p.lda = T; p.bsA = (long)S * T;
      p.Bw = tvv; p.ldb = (long)C_B * C_H; p.bsB = C_H;
      p.C = tctx; p.ldc = (long)C_B * C_H; p.bsC = C_H;
      p.M = S; p.Nn = C_H; p.K = T; p.scale = 1.f;
      launch_gemm<0, 0, 0>(p, C_B, st); }
    { GemmP p = mkG(tctx, C_H, PF(din, L + 6), C_H, PF(din, L + 7), tx, C_H,
                    (int)SB, C_H, C_H);
      p.Res = tx; p.ldres = C_H;
      launch_gemm<0, 0, 0>(p, 1, st); }
    run_ln(tx, PF(din, L + 14), PF(din, L + 15), nullptr, txn, C_H, 0, SB, st);
    { GemmP p = mkG(txn, C_H, PF(din, L + 10), C_FFN, PF(din, L + 11), tffh, C_FFN,
                    (int)SB, C_FFN, C_H);
      launch_gemm<1, 0, 0>(p, 1, st); }
    { GemmP p = mkG(tffh, C_FFN, PF(din, L + 12), C_H, PF(din, L + 13), tx, C_H,
                    (int)SB, C_H, C_FFN);
      p.Res = tx; p.ldres = C_H;
      launch_gemm<0, 0, 0>(p, 1, st); }
  }
  run_ln(tx, PF(din, tb + 32), PF(din, tb + 33), nullptr, out, C_H, 0, SB, st);
}

// ------------------------------------------------------------------- driver
extern "C" void kernel_launch(void* const* d_in, const int* in_sizes, int n_in,
                              void* d_out, int out_size, void* d_ws, size_t ws_size,
                              hipStream_t stream) {
  (void)in_sizes; (void)n_in; (void)out_size; (void)ws_size;
  hipStream_t st = stream;
  // leaf indices (depth-first insertion order of setup_inputs())
  const int L_EMB = 2, L_PROMPT = 3, L_ENCN = 4, L_ENCE = 9, L_PROC = 14;
  const int L_TR = 56, L_TRSELF = 90, L_TRSRC = 124;
  const int L_DECW = 158, L_DECB = 159, L_FCW = 160, L_FCB = 161;
  const int L_FCPW = 162, L_FCPB = 163, L_OUTW = 164, L_OUTB = 165;
  const int* rowI = PI(d_in, 166);
  const int* colI = PI(d_in, 166) + C_E;
  const int* sys  = PI(d_in, 168);

  // ---- workspace carve (floats)
  float* w = (float*)d_ws;
  auto alloc = [&](long n) { float* p = w; w += n; return p; };
  float* e     = alloc((long)C_E * C_H);
  float* ghid  = alloc((long)CH_EDGE * 2 * C_H);
  float* nhid  = alloc((long)C_N * 2 * C_H);
  float* agg   = alloc((long)C_N * C_H);
  float* cnt   = alloc(C_N);
  float* h     = alloc((long)C_N * C_H);
  float* ea    = alloc((long)C_E * 4);
  float* xd    = alloc((long)C_N * C_H);
  float* enIn  = alloc((long)C_NE * C_B * C_H);
  float* enOut = alloc((long)C_NE * C_B * C_H);
  float* gsum  = alloc((long)C_B * C_H);
  float* gvec  = alloc((long)C_B * C_H);
  float* catb  = alloc((long)C_NE * C_B * (2 * C_H + C_H2));
  float* dib   = alloc((long)C_NE * C_B * C_H);
  float* tx    = alloc(64L * C_B * C_H);
  float* txn   = alloc(64L * C_B * C_H);
  float* tq    = alloc(64L * C_B * C_H);
  float* tctx  = alloc(64L * C_B * C_H);
  float* tffh  = alloc(64L * C_B * C_FFN);
  float* tk    = alloc(256L * C_B * C_H);
  float* tv    = alloc(256L * C_B * C_H);
  float* tkn   = alloc(256L * C_B * C_H);
  float* tvn   = alloc(256L * C_B * C_H);
  float* tkk   = alloc(256L * C_B * C_H);
  float* tvv   = alloc(256L * C_B * C_H);
  float* tsc   = alloc((long)C_B * 64 * 256);

  // ---- edge features
  edge_attr_k<<<EL(C_E), 0, st>>>(PF(d_in, 1), ea, C_E);

  // ---- node encoder: prelu(x@w1+b1)@w2+b2 -> h
  { GemmP p = mkG(PF(d_in, 0), C_NATOM, PF(d_in, L_ENCN + 0), C_H,
                  PF(d_in, L_ENCN + 1), nhid, C_H, C_N, C_H, C_NATOM);
    p.aP = PF(d_in, L_ENCN + 2);
    launch_gemm<2, 0, 0>(p, 1, st); }
  { GemmP p = mkG(nhid, C_H, PF(d_in, L_ENCN + 3), C_H, PF(d_in, L_ENCN + 4),
                  h, C_H, C_N, C_H, C_H);
    launch_gemm<0, 0, 0>(p, 1, st); }

  // ---- edge encoder (chunked): prelu(ea@w1+b1)@w2+b2 -> e
  for (int c = 0; c < N_CHUNK; ++c) {
    long base = (long)c * CH_EDGE;
    { GemmP p = mkG(ea + base * 4, 4, PF(d_in, L_ENCE + 0), C_H,
                    PF(d_in, L_ENCE + 1), ghid, C_H, CH_EDGE, C_H, 4);
      p.aP = PF(d_in, L_ENCE + 2);
      launch_gemm<2, 0, 0>(p, 1, st); }
    { GemmP p = mkG(ghid, C_H, PF(d_in, L_ENCE + 3), C_H, PF(d_in, L_ENCE + 4),
                    e + base * C_H, C_H, CH_EDGE, C_H, C_H);
      launch_gemm<0, 0, 0>(p, 1, st); }
  }

  // ---- per-node in-degree counts (col is fixed across layers)
  fill_f32<<<EL(C_N), 0, st>>>(cnt, 0.f, C_N);
  count_k<<<EL(C_E), 0, st>>>(colI, cnt, C_E);

  // ---- GNN processor layers
  for (int l = 0; l < 3; ++l) {
    int LE = L_PROC + l * 14, LN_ = LE + 7;
    fill_f32<<<EL((long)C_N * C_H), 0, st>>>(agg, 0.f, (long)C_N * C_H);
    for (int c = 0; c < N_CHUNK; ++c) {
      int base = c * CH_EDGE;
      { GemmP p{};                                   // cat[h[row],h[col],e] @ w1
        p.Bw = PF(d_in, LE + 0); p.ldb = 2 * C_H; p.bias = PF(d_in, LE + 1);
        p.C = ghid; p.ldc = 2 * C_H;
        p.M = CH_EDGE; p.Nn = 2 * C_H; p.K = 3 * C_H; p.scale = 1.f;
        p.g0 = h; p.g2 = e; p.rowIdx = rowI; p.colIdx = colI; p.rowBase = base;
        launch_gemm<0, 1, 0>(p, 1, st); }
      run_ln(ghid, PF(d_in, LE + 2), PF(d_in, LE + 3), PF(d_in, LE + 4),
             ghid, 2 * C_H, 2, CH_EDGE, st);
      { GemmP p{};                                   // @ w2 ; e+=ne ; scatter agg
        p.A = ghid; p.lda = 2 * C_H;
        p.Bw = PF(d_in, LE + 5); p.ldb = C_H; p.bias = PF(d_in, LE + 6);
        p.M = CH_EDGE; p.Nn = C_H; p.K = 2 * C_H; p.scale = 1.f;
        p.eAcc = e; p.agg = agg; p.colIdx = colI; p.rowBase = base;
        launch_gemm<0, 0, 1>(p, 1, st); }
    }
    divide_k<<<EL((long)C_N * C_H), 0, st>>>(agg, cnt, (long)C_N * C_H);
    { GemmP p{};                                     // cat[h,agg] @ w1
      p.Bw = PF(d_in, LN_ + 0); p.ldb = 2 * C_H; p.bias = PF(d_in, LN_ + 1);
      p.C = nhid; p.ldc = 2 * C_H;
      p.M = C_N; p.Nn = 2 * C_H; p.K = 2 * C_H; p.scale = 1.f;
      p.g0 = h; p.g1 = agg;
      launch_gemm<0, 2, 0>(p, 1, st); }
    run_ln(nhid, PF(d_in, LN_ + 2), PF(d_in, LN_ + 3), PF(d_in, LN_ + 4),
           nhid, 2 * C_H, 2, C_N, st);
    { GemmP p = mkG(nhid, 2 * C_H, PF(d_in, LN_ + 5), C_H, PF(d_in, LN_ + 6),
                    h, C_H, C_N, C_H, 2 * C_H);
      p.Res = h; p.ldres = C_H;                      // h = h + nh
      launch_gemm<0, 0, 0>(p, 1, st); }
  }

  // ---- dense batch transpose + energy-query transformer
  xdense_k<<<EL((long)C_N * C_H), 0, st>>>(h, xd, (long)C_N * C_H);
  bcast_emb_k<<<EL((long)C_NE * C_B * C_H), 0, st>>>(PF(d_in, L_EMB), enIn,
                                                     (long)C_NE * C_B * C_H);
  run_transformer(d_in, L_TR, C_NE, C_NPG, enIn, xd, xd, enOut,
                  tx, txn, tq, tctx, tffh, tk, tv, tkn, tvn, tkk, tvv, tsc, st);

  // ---- graph decoder
  graphsum_k<<<dim3(C_B), dim3(C_H), 0, st>>>(h, gsum);
  { GemmP p = mkG(gsum, C_H, PF(d_in, L_DECW), C_H, PF(d_in, L_DECB),
                  gvec, C_H, C_B, C_H, C_H);
    launch_gemm<0, 0, 0>(p, 1, st); }

  float* dosG = (float*)d_out;
  float* hOut = (float*)d_out + (long)C_B * C_NE;
  float* dosS = hOut + (long)C_N * C_H;

  // ---- global DOS head
  cat_eg_k<<<EL((long)C_NE * C_B * 2 * C_H), 0, st>>>(enOut, gvec, catb,
                                                      (long)C_NE * C_B * 2 * C_H);
  { GemmP p = mkG(catb, 2 * C_H, PF(d_in, L_FCW), C_H, PF(d_in, L_FCB),
                  dib, C_H, C_NE * C_B, C_H, 2 * C_H);
    launch_gemm<3, 0, 0>(p, 1, st); }
  run_transformer(d_in, L_TRSELF, C_NE, C_NE, dib, dib, dib, dib,
                  tx, txn, tq, tctx, tffh, tk, tv, tkn, tvn, tkk, tvv, tsc, st);
  run_transformer(d_in, L_TRSRC, C_NE, C_NPG, dib, xd, xd, dib,
                  tx, txn, tq, tctx, tffh, tk, tv, tkn, tvn, tkk, tvv, tsc, st);
  outproj_k<<<EL(C_NE * C_B), 0, st>>>(dib, PF(d_in, L_OUTW), PF(d_in, L_OUTB), dosG);

  // ---- h output
  copy_f32<<<EL((long)C_N * C_H), 0, st>>>(h, hOut, (long)C_N * C_H);

  // ---- system-prompt DOS head
  cat_egp_k<<<EL((long)C_NE * C_B * (2 * C_H + C_H2)), 0, st>>>(
      enOut, gvec, PF(d_in, L_PROMPT), sys, catb,
      (long)C_NE * C_B * (2 * C_H + C_H2));
  { GemmP p = mkG(catb, 2 * C_H + C_H2, PF(d_in, L_FCPW), C_H, PF(d_in, L_FCPB),
                  dib, C_H, C_NE * C_B, C_H, 2 * C_H + C_H2);
    launch_gemm<3, 0, 0>(p, 1, st); }
  run_transformer(d_in, L_TRSELF, C_NE, C_NE, dib, dib, dib, dib,
                  tx, txn, tq, tctx, tffh, tk, tv, tkn, tvn, tkk, tvv, tsc, st);
  run_transformer(d_in, L_TRSRC, C_NE, C_NPG, dib, xd, xd, dib,
                  tx, txn, tq, tctx, tffh, tk, tv, tkn, tvn, tkk, tvv, tsc, st);
  outproj_k<<<EL(C_NE * C_B), 0, st>>>(dib, PF(d_in, L_OUTW), PF(d_in, L_OUTB), dosS);
}
